// PatchMatchOnce_14963666059435
// MI455X (gfx1250) — matrix operand order repaired and checked by reading, compiled, not run on hardware
//
#include <hip/hip_runtime.h>
#include <math.h>

// ---------------------------------------------------------------------------
// PatchMatch propagate+evaluate on MI455X (gfx1250, wave32).
//
// Strategy: features converted once to f16 (halves L2 gather traffic; both
// tables are L2-resident at 8MB each). One wave per pixel computes the
// 27x256 . 256x1 cost GEMV as two 16x16 WMMA accumulator tiles over 8 K=32
// steps using v_wmma_f32_16x16x32_f16 (f32 accumulate). Softmax + top-9 via
// wave32 shuffle reductions.
// ---------------------------------------------------------------------------

typedef _Float16 f16;
typedef f16   v16h __attribute__((ext_vector_type(16)));
typedef f16   v8h  __attribute__((ext_vector_type(8)));
typedef float v8f  __attribute__((ext_vector_type(8)));

union V16U { v16h v; v8h h[2]; };

#define NB     4
#define CCH    256
#define HDIM   64
#define WDIM   64
#define HWSZ   4096
#define NUMOFF 9
#define NCAND  27   // 3 * NUMOFF after propagate
#define TOPK   9    // NCAND / FILTER_SIZE

__device__ __forceinline__ int imin(int a, int b) { return a < b ? a : b; }
__device__ __forceinline__ int imax(int a, int b) { return a > b ? a : b; }

// ---------------------------------------------------------------------------
// Tiled transpose + f32 -> f16 convert.
// src[(bb*256 + r)*cols + c]  (row-major, 256 channel rows)
// dst[(bb*cols + c)*256 + r]  (pixel-major, channel-contiguous f16)
// block (32,8), grid (cols/32, 8, batch)
// ---------------------------------------------------------------------------
__global__ __launch_bounds__(256) void cvt_tr_kernel(
    const float* __restrict__ src, f16* __restrict__ dst, int cols)
{
  __shared__ float tile[32][33];
  const int bb = blockIdx.z;
  const int cBase = blockIdx.x * 32;
  const int rBase = blockIdx.y * 32;
#pragma unroll
  for (int i = 0; i < 4; ++i) {
    const int r = rBase + threadIdx.y + i * 8;
    tile[threadIdx.y + i * 8][threadIdx.x] =
        src[((size_t)(bb * CCH + r)) * cols + cBase + threadIdx.x];
  }
  __syncthreads();
#pragma unroll
  for (int i = 0; i < 4; ++i) {
    const int cc = cBase + threadIdx.y + i * 8;
    dst[((size_t)(bb * cols + cc)) * CCH + rBase + threadIdx.x] =
        (f16)tile[threadIdx.x][threadIdx.y + i * 8];
  }
}

// ---------------------------------------------------------------------------
// Fused propagate + evaluate. One wave32 per pixel.
// PASS==1: horizontal propagate from input float offsets; writes top-9
//          candidate indices (local py*W+px) to inds_out.
// PASS==2: vertical propagate from pass-1 indices; writes ox/oy/corr.
// ---------------------------------------------------------------------------
template <int PASS>
__global__ __launch_bounds__(256) void eval_kernel(
    const f16* __restrict__ Rt,     // (NB*HWSZ, 256) f16, gather table
    const f16* __restrict__ Lt,     // (NB*HWSZ, 256) f16, left features
    const float* __restrict__ offx, // pass1 only: (NB,9,H,W)
    const float* __restrict__ offy,
    const int* __restrict__ inds_in,  // pass2 only: (NB,9,H,W)
    int* __restrict__ inds_out,       // pass1 only
    float* __restrict__ out_ox,       // pass2 only
    float* __restrict__ out_oy,
    float* __restrict__ out_corr)
{
  __shared__ float scost[8][32];
  const int lane = threadIdx.x & 31;
  const int wid  = threadIdx.x >> 5;
  const int pix  = blockIdx.x * 8 + wid;      // 0..16383
  const int b = pix >> 12;
  const int p = pix & (HWSZ - 1);
  const int y = p >> 6;
  const int x = p & 63;

  // Candidate index for candidate m in [0,27): j = m%9 offset channel,
  // s = m/9 in {0,1,2} = {from prev, self, from next} along x (pass1) /
  // y (pass2). Out-of-range source => offset 0 (matches jnp.pad).
  auto candInd = [&](int m) -> int {
    const int j = m % NUMOFF;
    const int s = m / NUMOFF;
    if (PASS == 1) {
      const int xs = x + s - 1;
      float ox = 0.f, oy = 0.f;
      if (xs >= 0 && xs < WDIM) {
        const int o = ((b * NUMOFF + j) << 12) + (y << 6) + xs;
        ox = offx[o];
        oy = offy[o];
      }
      const float px = fminf(fmaxf((float)x + ox, 0.f), 63.f);
      const float py = fminf(fmaxf((float)y + oy, 0.f), 63.f);
      return (int)(py * 64.f + px);
    } else {
      const int ys = y + s - 1;
      int ox = 0, oy = 0;
      if (ys >= 0 && ys < HDIM) {
        const int ind = inds_in[((b * NUMOFF + j) << 12) + (ys << 6) + x];
        ox = (ind & 63) - x;
        oy = (ind >> 6) - ys;
      }
      const int px = imin(imax(x + ox, 0), 63);
      const int py = imin(imax(y + oy, 0), 63);
      return (py << 6) + px;
    }
  };

  // This lane supplies A-matrix rows m0 (tile0) and m1 (tile1).
  const int m0 = lane & 15;
  const int m1 = m0 + 16;
  const int ind0 = candInd(m0);
  const int ind1 = (m1 < NCAND) ? candInd(m1) : 0;  // pad rows gather col 0

  const f16* r0 = Rt + (((size_t)(b << 12) + (size_t)ind0) << 8);
  const f16* r1 = Rt + (((size_t)(b << 12) + (size_t)ind1) << 8);
  const f16* lf = Lt + ((size_t)pix << 8);
  const int hi = lane >> 4;  // which half-wave: selects K sub-ranges

  // ISA f16 A 16x32 layout: lanes 0-15 hold K{0..7,16..23}, lanes 16-31 hold
  // K{8..15,24..31}. B 32x16: same K order as A
  // (broadcast identical across N columns).
  v8f acc0 = {};
  v8f acc1 = {};
#pragma unroll
  for (int kb = 0; kb < CCH; kb += 32) {
    V16U a0, a1, bm;
    const int klo = kb + hi * 8;
    const int khi = kb + 16 + hi * 8;
    a0.h[0] = *(const v8h*)(r0 + klo);
    a0.h[1] = *(const v8h*)(r0 + khi);
    a1.h[0] = *(const v8h*)(r1 + klo);
    a1.h[1] = *(const v8h*)(r1 + khi);
    const int kbB = kb + hi * 8;
    bm.h[0] = *(const v8h*)(lf + kbB);
    bm.h[1] = *(const v8h*)(lf + kbB + 16);
    acc0 = __builtin_amdgcn_wmma_f32_16x16x32_f16(false, a0.v, false, bm.v,
                                                  (short)0, acc0, false, false);
    acc1 = __builtin_amdgcn_wmma_f32_16x16x32_f16(false, a1.v, false, bm.v,
                                                  (short)0, acc1, false, false);
  }

  // D layout: lane n<16 VGPR r = M=r ; lane n>=16 VGPR r = M=r+8. All N
  // columns identical (B was broadcast), so lanes 0 and 16 hold everything.
  float* Lc = scost[wid];
#pragma unroll
  for (int r = 0; r < 8; ++r) {
    if (lane == 0)  { Lc[r]      = acc0[r]; Lc[16 + r] = acc1[r]; }
    if (lane == 16) { Lc[8 + r]  = acc0[r]; Lc[24 + r] = acc1[r]; }
  }
  __syncthreads();

  // Softmax over 27 candidates (cost / T, T = 0.01), lane n owns candidate n.
  const float v = (lane < NCAND) ? Lc[lane] * 100.0f : -3.0e38f;
  float mx = v;
#pragma unroll
  for (int o = 16; o >= 1; o >>= 1) mx = fmaxf(mx, __shfl_xor(mx, o, 32));
  const float e = (lane < NCAND) ? __expf(v - mx) : 0.0f;
  float sum = e;
#pragma unroll
  for (int o = 16; o >= 1; o >>= 1) sum += __shfl_xor(sum, o, 32);
  const float myp = e / sum;
  // Lane n's candidate index (lanes 16..26 computed candidate n as ind1).
  const int myInd = (lane < 16) ? ind0 : ind1;

  // Top-9 by iterative wave-argmax (ties -> lowest lane, as jax top_k).
  float pw = myp;
#pragma unroll
  for (int k = 0; k < TOPK; ++k) {
    float m = pw;
#pragma unroll
    for (int o = 16; o >= 1; o >>= 1) m = fmaxf(m, __shfl_xor(m, o, 32));
    const unsigned long long bal = __ballot(pw == m);
    const int win = __ffsll(bal) - 1;
    if (lane == win) {
      const size_t o = ((size_t)(b * NUMOFF + k) << 12) + (size_t)p;
      if (PASS == 1) {
        inds_out[o] = myInd;
      } else {
        out_ox[o]   = (float)((myInd & 63) - x);
        out_oy[o]   = (float)((myInd >> 6) - y);
        out_corr[o] = pw;
      }
      pw = -1.0f;
    }
  }
}

// ---------------------------------------------------------------------------
extern "C" void kernel_launch(void* const* d_in, const int* in_sizes, int n_in,
                              void* d_out, int out_size, void* d_ws,
                              size_t ws_size, hipStream_t stream)
{
  (void)in_sizes; (void)n_in; (void)out_size; (void)ws_size;

  const float* left  = (const float*)d_in[0];  // (4,256,4096)
  const float* right = (const float*)d_in[1];  // (256,16384)
  const float* offx  = (const float*)d_in[2];  // (4,9,64,64)
  const float* offy  = (const float*)d_in[3];

  char* ws = (char*)d_ws;
  f16* Rt    = (f16*)ws;                                      // 8 MB
  f16* Lt    = (f16*)(ws + (size_t)16384 * 256 * 2);          // 8 MB
  int* inds1 = (int*)(ws + (size_t)16384 * 256 * 4);          // 576 KB

  float* out_ox   = (float*)d_out;                 // (4,9,64,64)
  float* out_oy   = out_ox + NB * NUMOFF * HWSZ;   // (4,9,64,64)
  float* out_corr = out_oy + NB * NUMOFF * HWSZ;   // (4,9,4096)

  const dim3 blk(32, 8, 1);
  // right: 256 x 16384 -> (16384, 256) f16
  cvt_tr_kernel<<<dim3(16384 / 32, CCH / 32, 1), blk, 0, stream>>>(right, Rt, 16384);
  // left: per-batch 256 x 4096 -> (b*4096 + p, 256) f16
  cvt_tr_kernel<<<dim3(HWSZ / 32, CCH / 32, NB), blk, 0, stream>>>(left, Lt, HWSZ);

  // Pass 1: horizontal propagate + evaluate -> top-9 indices into ws.
  eval_kernel<1><<<NB * HWSZ / 8, 256, 0, stream>>>(
      Rt, Lt, offx, offy, nullptr, inds1, nullptr, nullptr, nullptr);
  // Pass 2: vertical propagate + evaluate -> ox, oy, corr.
  eval_kernel<2><<<NB * HWSZ / 8, 256, 0, stream>>>(
      Rt, Lt, nullptr, nullptr, inds1, nullptr, out_ox, out_oy, out_corr);
}
